// LSTM_55989193670959
// MI455X (gfx1250) — matrix operand order repaired and checked by reading, not compiled, not run on hardware
//
#include <hip/hip_runtime.h>
#include <hip/hip_bf16.h>

typedef __bf16 bf16_t;
typedef __attribute__((ext_vector_type(16))) __bf16 v16bf;
typedef __attribute__((ext_vector_type(8)))  __bf16 v8bf;
typedef __attribute__((ext_vector_type(8)))  float  v8f;

#define LSTM_B   64
#define LSTM_S   512
#define LSTM_D   512
#define LSTM_G4  2048
#define REC_NWG  8
#define REC_THR  128

union AFrag { v16bf v; v8bf h[2]; };

__device__ __forceinline__ float sigmoidf_fast(float x) {
  float e = __builtin_amdgcn_exp2f(x * -1.44269504088896340736f);
  return __builtin_amdgcn_rcpf(1.0f + e);
}

__device__ __forceinline__ float tanhf_fast(float x) {
#if __has_builtin(__builtin_amdgcn_tanhf)
  return __builtin_amdgcn_tanhf(x);
#else
  float xx = fminf(fmaxf(x, -15.0f), 15.0f);
  float e  = __builtin_amdgcn_exp2f(xx * 2.88539008177792681472f); // e^(2x)
  return (e - 1.0f) * __builtin_amdgcn_rcpf(e + 1.0f);
#endif
}

// CDNA5 async DMA: global -> LDS, 16B per lane, tracked by ASYNCcnt (no VGPR data path)
__device__ __forceinline__ void async_copy16(unsigned lds_off, const void* gsrc) {
  asm volatile("global_load_async_to_lds_b128 %0, %1, off"
               :: "v"(lds_off), "v"(gsrc) : "memory");
}
__device__ __forceinline__ void wait_async0() {
  asm volatile("s_wait_asynccnt 0x0" ::: "memory");
}
__device__ __forceinline__ void wait_async_le2() {
  asm volatile("s_wait_asynccnt 0x2" ::: "memory");
}

// ---- pack 4 gate weight matrices [D][H] f32 -> N-major bf16 [4H][D] ----
__global__ __launch_bounds__(256) void pack_weights(
    const float* __restrict__ w0, const float* __restrict__ w1,
    const float* __restrict__ w2, const float* __restrict__ w3,
    bf16_t* __restrict__ out) {
  int idx = blockIdx.x * blockDim.x + threadIdx.x;   // over 2048*512
  int g = idx >> 9;
  int d = idx & 511;
  const float* w = (g < 512) ? w0 : (g < 1024) ? w1 : (g < 1536) ? w2 : w3;
  out[idx] = (bf16_t)w[d * 512 + (g & 511)];
}

// ---- inputs [B][S][D] f32 -> Xr bf16 [S*B][D] (row = s*64+b) ----
__global__ __launch_bounds__(256) void pack_x(
    const float* __restrict__ x, bf16_t* __restrict__ out) {
  int idx = blockIdx.x * blockDim.x + threadIdx.x;   // over 64*512*512
  int d = idx & 511;
  int row = idx >> 9;           // s*64 + b
  int b = row & 63, s = row >> 6;
  out[idx] = (bf16_t)x[((size_t)(b << 9) + s) * 512 + d];
}

__global__ __launch_bounds__(256) void init_state(
    bf16_t* __restrict__ hbuf0, unsigned* __restrict__ cnt) {
  int idx = blockIdx.x * blockDim.x + threadIdx.x;   // 32768 total
  hbuf0[idx] = (bf16_t)0.0f;
  if (idx == 0) *cnt = 0u;
}

// ---- Phase 1: xproj[(s*64+b)][g] = Xr @ W + bias (bf16 WMMA, f32 acc) ----
// B tile (128 cols x 32 K = 8 KB) is shared by all 8 waves: async-DMA'd into a
// 2-deep LDS pipeline; WMMAs consume buffer t while the async engine fills t+1.
__global__ __launch_bounds__(256) void xproj_gemm(
    const bf16_t* __restrict__ Xr,    // [32768][512] bf16
    const bf16_t* __restrict__ Wt,    // [2048][512]  bf16 (N-major)
    const float* __restrict__ b0, const float* __restrict__ b1,
    const float* __restrict__ b2, const float* __restrict__ b3,
    float* __restrict__ xproj)        // [32768][2048] f32
{
  __shared__ bf16_t Blds[2 * 8 * 512];   // 16 KB double buffer

  const int tid  = threadIdx.x;
  const int lane = tid & 31;
  const int wid  = tid >> 5;   // 0..7 -> M sub-tile
  const int ln   = lane & 15;
  const int hf   = lane >> 4;

  const int Nbase = (blockIdx.x & 15) * 128;
  const int Mbase = (blockIdx.x >> 4) * 128;

  const bf16_t* arow = Xr + (size_t)(Mbase + wid * 16 + ln) * 512;

  // async fill of one B buffer: 512 x 16B pieces, exactly 2 per thread.
  // piece q = (nt*2 + p)*32 + l ; LDS elem = buf*4096 + nt*512 + p*256 + l*8
  auto fillB = [&](int buf, int k0) {
#pragma unroll
    for (int qq = 0; qq < 2; ++qq) {
      int q = tid + qq * 256;
      int l = q & 31, p = (q >> 5) & 1, nt = q >> 6;
      int lln = l & 15, lhf = l >> 4;
      const bf16_t* src = Wt + (size_t)(Nbase + nt * 16 + lln) * 512
                             + k0 + 8 * lhf + 16 * p;
      async_copy16((unsigned)(size_t)(Blds + buf * 4096 + nt * 512 + p * 256 + l * 8),
                   src);
    }
  };
  auto loadA = [&](AFrag& a, int k0) {
    a.h[0] = *(const v8bf*)(arow + k0 + 8 * hf);
    a.h[1] = *(const v8bf*)(arow + k0 + 16 + 8 * hf);
  };

  v8f acc[8] = {};
  AFrag a0, a1;
  fillB(0, 0);
  loadA(a0, 0);

#pragma unroll
  for (int t = 0; t < 16; ++t) {
    const int k0 = t * 32;
    if (t + 1 < 16) {
      fillB((t + 1) & 1, k0 + 32);   // async prefetch next B tile
      loadA(a1, k0 + 32);            // next A fragment into registers
      wait_async_le2();              // previous fill done, new one in flight
    } else {
      wait_async0();
    }
    __syncthreads();                 // all threads' fills of buf[t&1] visible

    const bf16_t* bbase = Blds + (t & 1) * 4096;
#pragma unroll
    for (int nt = 0; nt < 8; ++nt) {
      AFrag b;
      b.h[0] = *(const v8bf*)(bbase + nt * 512 + lane * 8);
      b.h[1] = *(const v8bf*)(bbase + nt * 512 + 256 + lane * 8);
      acc[nt] = __builtin_amdgcn_wmma_f32_16x16x32_bf16(
          false, a0.v, false, b.v, (short)0, acc[nt], false, false);
    }
    __syncthreads();                 // readers done before buffer is refilled
    a0 = a1;
  }

#pragma unroll
  for (int nt = 0; nt < 8; ++nt) {
    int col  = Nbase + nt * 16 + ln;
    int gate = col >> 9;
    const float* bp = (gate == 0) ? b0 : (gate == 1) ? b1 : (gate == 2) ? b2 : b3;
    float bias = bp[col & 511];
#pragma unroll
    for (int e = 0; e < 8; ++e) {
      int m = Mbase + wid * 16 + e + 8 * hf;
      xproj[(size_t)m * 2048 + col] = acc[nt][e] + bias;
    }
  }
}

// ---- Phase 2: 8 persistent WGs x 4 waves; U slice (256KB) LDS-resident ----
// Wave `wid` owns H-columns [wg*64 + wid*16, +16) for all 4 gates and all 64
// batch rows: 16 C tiles, so each LDS B fragment feeds 4 WMMAs.
// LDS fragment layout: frag f = wid*64 + g*16 + t occupies 512 bf16;
// piece p (0/1), lane l at element (f*512 + p*256 + l*8) -> lane-contiguous b128.
__global__ __launch_bounds__(128) void lstm_recur(
    const float* __restrict__ xproj,  // [512][64][2048] f32
    const bf16_t* __restrict__ Ut,    // [2048][512] bf16 (N-major)
    bf16_t* __restrict__ hbuf,        // [2][64][512] bf16 (double buffer)
    float* __restrict__ out,          // [64][512] f32
    unsigned* __restrict__ cnt)
{
  extern __shared__ char smem[];
  bf16_t* Ulds = (bf16_t*)smem;       // 256 KB

  const int tid  = threadIdx.x;
  const int lane = tid & 31;
  const int wid  = tid >> 5;          // 0..3 = ntile
  const int ln   = lane & 15;
  const int hf   = lane >> 4;
  const int wg   = blockIdx.x;        // 0..7, owns H cols [wg*64, wg*64+64)

  // one-time async DMA of this WG's U slice into LDS (16384 x 16B pieces)
  for (int q = tid; q < 256 * 64; q += REC_THR) {
    int l = q & 31, p = (q >> 5) & 1, f = q >> 6;
    int t = f & 15, g = (f >> 4) & 3, nt = f >> 6;
    int lln = l & 15, lhf = l >> 4;
    const bf16_t* src = Ut + (size_t)(g * 512 + wg * 64 + nt * 16 + lln) * 512
                           + t * 32 + 8 * lhf + 16 * p;
    async_copy16((unsigned)(size_t)(Ulds + (size_t)q * 8), src);
  }
  wait_async0();
  __syncthreads();

  const int hcol = wg * 64 + wid * 16 + ln;

  v8f c[4] = {};   // cell state per M-tile, VGPR-resident for all 512 steps

  for (int s = 0; s < LSTM_S; ++s) {
    const bf16_t* hread = hbuf + (size_t)(s & 1) * (LSTM_B * LSTM_D);

    // seed accumulators with x_proj (fuses the "+ xp" add into the MMA chain)
    v8f acc[4][4];   // [gate][mtile]
    const float* xp = xproj + (size_t)s * (LSTM_B * LSTM_G4);
#pragma unroll
    for (int mt = 0; mt < 4; ++mt)
#pragma unroll
      for (int e = 0; e < 8; ++e) {
        const float* row = xp + (size_t)(mt * 16 + e + 8 * hf) * 2048;
        acc[0][mt][e] = row[0 * 512 + hcol];
        acc[1][mt][e] = row[1 * 512 + hcol];
        acc[2][mt][e] = row[2 * 512 + hcol];
        acc[3][mt][e] = row[3 * 512 + hcol];
      }

#pragma unroll 2
    for (int t = 0; t < 16; ++t) {
      const int k0 = t * 32;
      AFrag a[4];
#pragma unroll
      for (int mt = 0; mt < 4; ++mt) {
        const bf16_t* ar = hread + (size_t)(mt * 16 + ln) * 512 + k0 + 8 * hf;
        a[mt].h[0] = *(const v8bf*)(ar);
        a[mt].h[1] = *(const v8bf*)(ar + 16);
      }
#pragma unroll
      for (int g = 0; g < 4; ++g) {
        const bf16_t* bp = Ulds + (size_t)(wid * 64 + g * 16 + t) * 512;
        AFrag b;
        b.h[0] = *(const v8bf*)(bp + lane * 8);          // piece 0
        b.h[1] = *(const v8bf*)(bp + 256 + lane * 8);    // piece 1
#pragma unroll
        for (int mt = 0; mt < 4; ++mt)
          acc[g][mt] = __builtin_amdgcn_wmma_f32_16x16x32_bf16(
              false, a[mt].v, false, b.v, (short)0, acc[g][mt], false, false);
      }
    }

    bf16_t* hw = hbuf + (size_t)((s + 1) & 1) * (LSTM_B * LSTM_D);
#pragma unroll
    for (int mt = 0; mt < 4; ++mt)
#pragma unroll
      for (int e = 0; e < 8; ++e) {
        float i_g = sigmoidf_fast(acc[0][mt][e]);
        float f_g = sigmoidf_fast(acc[1][mt][e]);
        float o_g = sigmoidf_fast(acc[2][mt][e]);
        float c_t = tanhf_fast(acc[3][mt][e]);
        float cn  = f_g * c[mt][e] + i_g * c_t;
        c[mt][e] = cn;
        float hn = o_g * tanhf_fast(cn);
        int m = mt * 16 + e + 8 * hf;
        hw[(size_t)m * 512 + hcol] = (bf16_t)hn;
        if (s == LSTM_S - 1) out[(size_t)m * 512 + hcol] = hn;
      }

    // grid barrier across the 8 persistent WGs
    __syncthreads();
    if (tid == 0) {
      __threadfence();
      __hip_atomic_fetch_add(cnt, 1u, __ATOMIC_RELEASE, __HIP_MEMORY_SCOPE_AGENT);
      unsigned target = (unsigned)REC_NWG * (unsigned)(s + 1);
      while (__hip_atomic_load(cnt, __ATOMIC_ACQUIRE, __HIP_MEMORY_SCOPE_AGENT) < target)
        __builtin_amdgcn_s_sleep(1);
    }
    __syncthreads();
    __threadfence();
  }
}

extern "C" void kernel_launch(void* const* d_in, const int* in_sizes, int n_in,
                              void* d_out, int out_size, void* d_ws, size_t ws_size,
                              hipStream_t stream) {
  const float* inputs = (const float*)d_in[0];
  const float* Wg[4]  = {(const float*)d_in[1], (const float*)d_in[2],
                         (const float*)d_in[3], (const float*)d_in[4]};
  const float* Ug[4]  = {(const float*)d_in[5], (const float*)d_in[6],
                         (const float*)d_in[7], (const float*)d_in[8]};
  const float* bg[4]  = {(const float*)d_in[9], (const float*)d_in[10],
                         (const float*)d_in[11], (const float*)d_in[12]};

  char* ws = (char*)d_ws;
  size_t off = 0;
  bf16_t* Wt = (bf16_t*)(ws + off); off += (size_t)LSTM_G4 * LSTM_D * 2;          // 2 MB
  bf16_t* Ut = (bf16_t*)(ws + off); off += (size_t)LSTM_G4 * LSTM_D * 2;          // 2 MB
  bf16_t* Xr = (bf16_t*)(ws + off); off += (size_t)LSTM_S * LSTM_B * LSTM_D * 2;  // 32 MB
  float* xproj = (float*)(ws + off); off += (size_t)LSTM_S * LSTM_B * LSTM_G4 * 4;// 256 MB
  bf16_t* hbuf = (bf16_t*)(ws + off); off += (size_t)2 * LSTM_B * LSTM_D * 2;     // 128 KB
  unsigned* cnt = (unsigned*)(ws + off); off += 256;

  pack_weights<<<4096, 256, 0, stream>>>(Wg[0], Wg[1], Wg[2], Wg[3], Wt);
  pack_weights<<<4096, 256, 0, stream>>>(Ug[0], Ug[1], Ug[2], Ug[3], Ut);
  pack_x<<<65536, 256, 0, stream>>>(inputs, Xr);
  init_state<<<128, 256, 0, stream>>>(hbuf, cnt);

  xproj_gemm<<<4096, 256, 0, stream>>>(Xr, Wt, bg[0], bg[1], bg[2], bg[3], xproj);
  lstm_recur<<<REC_NWG, REC_THR, 262144, stream>>>(xproj, Ut, hbuf, (float*)d_out, cnt);
}
